// CrossModalAttention_65017214927321
// MI455X (gfx1250) — matrix operand order repaired and checked by reading, not compiled, not run on hardware
//
#include <hip/hip_runtime.h>
#include <hip/hip_bf16.h>
#include <stdint.h>

// ---------------------------------------------------------------------------
// Transformer encoder block for MI455X (gfx1250), bf16 WMMA everywhere.
//   B=4, S=2048, E=512, H=8, Dh=64, FFN=2048
// Round 4: async global->LDS double-buffered GEMM; ping-pong buffers are now
//          computed per-iteration (no LDS-derived pointer arrays -> no
//          addrspacecast static initializers for ld.lld to choke on).
// ---------------------------------------------------------------------------

#define EMB   512
#define NH    8
#define DH    64
#define FFN   2048
#define BATCH 4
#define SEQ   2048
#define MROWS (BATCH * SEQ)   // 8192

typedef __attribute__((ext_vector_type(16))) __bf16       v16bf;
typedef __attribute__((ext_vector_type(8)))  float        v8f;
typedef __attribute__((ext_vector_type(4)))  unsigned int u32x4;
typedef __attribute__((ext_vector_type(4)))  int          i32x4;
typedef __attribute__((address_space(1)))    i32x4        i32x4_g;   // global
typedef __attribute__((address_space(3)))    i32x4        i32x4_l;   // LDS

union Frag {            // one WMMA 16-bit A/B operand: 16 bf16 per lane
    u32x4 u[2];
    v16bf v;
};

__device__ __forceinline__ v8f wmma_bf16(const v16bf& a, const v16bf& b, const v8f& c) {
    return __builtin_amdgcn_wmma_f32_16x16x32_bf16(false, a, false, b, (short)0, c,
                                                   false, false);
}

// ---- CDNA5 async global->LDS path (GLOBAL_LOAD_ASYNC_TO_LDS_B128) ---------
#define HAS_ASYNC_LDS __has_builtin(__builtin_amdgcn_global_load_async_to_lds_b128)

__device__ __forceinline__ void async_copy16(const void* gsrc, void* ldst) {
#if HAS_ASYNC_LDS
    __builtin_amdgcn_global_load_async_to_lds_b128((i32x4_g*)gsrc, (i32x4_l*)ldst, 0, 0);
#else
    *(u32x4*)ldst = *(const u32x4*)gsrc;
#endif
}

__device__ __forceinline__ void wait_async0() {
#if HAS_ASYNC_LDS
#if __has_builtin(__builtin_amdgcn_s_wait_asynccnt)
    __builtin_amdgcn_s_wait_asynccnt(0);
#else
    asm volatile("s_wait_asynccnt 0x0" ::: "memory");
#endif
#endif
}

// ---------------------------------------------------------------------------
// fp32 -> bf16 conversion (grid-stride)
// ---------------------------------------------------------------------------
__global__ void cvt_bf16_kernel(const float* __restrict__ s, __bf16* __restrict__ d, int n) {
    int i = blockIdx.x * blockDim.x + threadIdx.x;
    int stride = gridDim.x * blockDim.x;
    for (; i < n; i += stride) d[i] = (__bf16)s[i];
}

// W[K][N] fp32  ->  Wt[N][K] bf16  (weights are tiny; simplicity wins)
__global__ void cvtT_bf16_kernel(const float* __restrict__ W, __bf16* __restrict__ Wt,
                                 int K, int N) {
    int i = blockIdx.x * blockDim.x + threadIdx.x;
    int n = K * N;
    int stride = gridDim.x * blockDim.x;
    for (; i < n; i += stride) {
        int k = i / N, c = i - k * N;
        Wt[(size_t)c * K + k] = (__bf16)W[i];
    }
}

// ---------------------------------------------------------------------------
// GEMM: D[M,N] = A[M,K](bf16) * Bt[N,K](bf16)^T + bias, templated epilogue.
// Block tile 128x128, 8 waves (2 N x 4 M), wave tile 32x64, K-step 32.
// Double-buffered LDS, filled by async global->LDS DMA (ASYNCcnt-tracked).
// ---------------------------------------------------------------------------
enum { EP_QK = 0, EP_V = 1, EP_O = 2, EP_F1 = 3, EP_F2 = 4 };

#define LSTR 40                             // bf16 stride -> 80B rows (16B aligned)
#define TILEB ((size_t)128 * LSTR * 2)      // bytes per 128x32 bf16 tile

__device__ __forceinline__ void stage_tile(const __bf16* __restrict__ A,
                                           const __bf16* __restrict__ Bt,
                                           __bf16* la, __bf16* lb,
                                           int m0, int n0, int K, int k0, int tid) {
#pragma unroll
    for (int it = 0; it < 2; ++it) {
        int idx = tid + it * 256;  // 0..511 chunks of 16B
        int row = idx >> 2;
        int cc  = idx & 3;
        async_copy16(A + (size_t)(m0 + row) * K + k0 + cc * 8, la + row * LSTR + cc * 8);
        async_copy16(Bt + (size_t)(n0 + row) * K + k0 + cc * 8, lb + row * LSTR + cc * 8);
    }
}

template <int MODE>
__global__ __launch_bounds__(256) void gemm_kernel(
    const __bf16* __restrict__ A, const __bf16* __restrict__ Bt,
    const float* __restrict__ bias, void* __restrict__ out,
    const float* __restrict__ resid, int Mtot, int Ntot, int K) {
    const int tid  = threadIdx.x;
    const int lane = tid & 31;
    const int wave = tid >> 5;
    const int wm   = wave >> 1;  // 0..3 (M)
    const int wn   = wave & 1;   // 0..1 (N)
    const int m0   = blockIdx.y * 128;
    const int n0   = blockIdx.x * 128;

    extern __shared__ char smem[];
    // buffer layout: [la0][lb0][la1][lb1]; pointers derived per-iteration

    v8f acc[2][4];
#pragma unroll
    for (int i = 0; i < 2; ++i)
#pragma unroll
        for (int j = 0; j < 4; ++j)
#pragma unroll
            for (int e = 0; e < 8; ++e) acc[i][j][e] = 0.0f;

    const int rsel  = (lane & 16) ? (lane - 16) : lane;
    const int koffA = (lane & 16) ? 8 : 0;   // A: two 8-elem chunks (+0,+16)
    const int koffB = (lane & 16) ? 8 : 0;   // B: two 8-elem chunks (+0,+16)

    // prologue: fill buffer 0
    stage_tile(A, Bt, (__bf16*)smem, (__bf16*)(smem + TILEB), m0, n0, K, 0, tid);

    for (int k0 = 0; k0 < K; k0 += 32) {
        const int cur = (k0 >> 5) & 1;
        __bf16* la = (__bf16*)(smem + (size_t)cur * 2 * TILEB);
        __bf16* lb = (__bf16*)(smem + (size_t)cur * 2 * TILEB + TILEB);

        wait_async0();          // my async fills for buf[cur] landed in LDS
        __syncthreads();        // everyone's fills landed
        if (k0 + 32 < K) {      // kick the next panel while we compute
            __bf16* lan = (__bf16*)(smem + (size_t)(cur ^ 1) * 2 * TILEB);
            __bf16* lbn = (__bf16*)(smem + (size_t)(cur ^ 1) * 2 * TILEB + TILEB);
            stage_tile(A, Bt, lan, lbn, m0, n0, K, k0 + 32, tid);
        }

        Frag fa[2], fb[4];
#pragma unroll
        for (int mt = 0; mt < 2; ++mt) {
            const __bf16* p = la + (wm * 32 + mt * 16 + rsel) * LSTR + koffA;
            fa[mt].u[0] = *(const u32x4*)(p);
            fa[mt].u[1] = *(const u32x4*)(p + 16);
        }
#pragma unroll
        for (int nt = 0; nt < 4; ++nt) {
            const __bf16* p = lb + (wn * 64 + nt * 16 + rsel) * LSTR + koffB;
            fb[nt].u[0] = *(const u32x4*)(p);
            fb[nt].u[1] = *(const u32x4*)(p + 16);
        }
#pragma unroll
        for (int mt = 0; mt < 2; ++mt)
#pragma unroll
            for (int nt = 0; nt < 4; ++nt)
                acc[mt][nt] = wmma_bf16(fa[mt].v, fb[nt].v, acc[mt][nt]);
        __syncthreads();        // done reading buf[cur]; next iter may refill it
    }

    // ---- epilogue ----
    const int nll   = lane & 15;
    const int mhalf = (lane & 16) ? 8 : 0;
#pragma unroll
    for (int mt = 0; mt < 2; ++mt)
#pragma unroll
        for (int nt = 0; nt < 4; ++nt) {
            const int n  = n0 + wn * 64 + nt * 16 + nll;
            const float bv = bias[n];
#pragma unroll
            for (int i = 0; i < 8; ++i) {
                const int m = m0 + wm * 32 + mt * 16 + i + mhalf;
                float v = acc[mt][nt][i] + bv;
                if constexpr (MODE == EP_QK) {
                    __bf16* o = (__bf16*)out;
                    int b = m >> 11, srow = m & (SEQ - 1);
                    int h = n >> 6, d = n & 63;
                    o[(((size_t)b * NH + h) * SEQ + srow) * DH + d] = (__bf16)v;
                } else if constexpr (MODE == EP_V) {
                    __bf16* o = (__bf16*)out;
                    int b = m >> 11, srow = m & (SEQ - 1);
                    int h = n >> 6, d = n & 63;
                    o[(((size_t)b * NH + h) * DH + d) * SEQ + srow] = (__bf16)v;
                } else if constexpr (MODE == EP_O || MODE == EP_F2) {
                    float* o = (float*)out;
                    o[(size_t)m * Ntot + n] = v + resid[(size_t)m * Ntot + n];
                } else {  // EP_F1: exact GELU, bf16 out
                    __bf16* o = (__bf16*)out;
                    float g = 0.5f * v * (1.0f + erff(v * 0.70710678118654752f));
                    o[(size_t)m * Ntot + n] = (__bf16)g;
                }
            }
        }
}

// ---------------------------------------------------------------------------
// Attention: one block = (b, h, 64 query rows); 4 waves, 16 q-rows each.
// Phase 1: S = (Q Kt)/8 via WMMA, bf16 scores parked in LDS, row max tracked.
// Phase 2: P = exp(S-m) rebuilt in A-fragment layout from LDS, O += P V via
//          WMMA against transposed V, normalize by row sums.
// ---------------------------------------------------------------------------
__global__ __launch_bounds__(128) void attention_kernel(
    const __bf16* __restrict__ Q,   // [B,H,S,DH]
    const __bf16* __restrict__ Km,  // [B,H,S,DH]
    const __bf16* __restrict__ Vt,  // [B,H,DH,S]
    __bf16* __restrict__ ctx) {     // [B*S, EMB] col = h*64+d
    const int tid  = threadIdx.x;
    const int lane = tid & 31;
    const int wave = tid >> 5;
    const int blk  = blockIdx.x;
    const int qb   = blk & ((SEQ / 64) - 1);
    const int bh   = blk >> 5;  // SEQ/64 == 32
    const int b    = bh >> 3, h = bh & 7;

    extern __shared__ char smem[];
    const int SC_STR = SEQ + 8;  // 2056 bf16 -> 4112B rows (16B aligned)
    const int QSTR   = 72;       // 144B rows
    __bf16* sc     = (__bf16*)smem;                                  // 64 x 2056
    __bf16* lq     = (__bf16*)(smem + 64 * SC_STR * 2);              // 64 x 72
    float*  rowmax = (float*)(smem + 64 * SC_STR * 2 + 64 * QSTR * 2);
    float*  rowsum = rowmax + 64;

    // ---- stage the 64x64 Q tile asynchronously; init stats ----
#pragma unroll
    for (int it = 0; it < 4; ++it) {
        int idx = tid + it * 128;  // 0..511 chunks of 16B
        int row = idx >> 3;
        int cc  = idx & 7;
        async_copy16(Q + ((size_t)bh * SEQ + qb * 64 + row) * DH + cc * 8,
                     lq + row * QSTR + cc * 8);
    }
    if (tid < 64) { rowmax[tid] = -1e30f; rowsum[tid] = 0.0f; }
    wait_async0();
    __syncthreads();

    const int qrb   = wave * 16;
    const int rsel  = (lane & 16) ? (lane - 16) : lane;
    const int koffA = (lane & 16) ? 8 : 0;
    const int koffB = (lane & 16) ? 8 : 0;
    const int mhalf = (lane & 16) ? 8 : 0;

    Frag aq[2];
#pragma unroll
    for (int ks = 0; ks < 2; ++ks) {
        const __bf16* p = lq + (qrb + rsel) * QSTR + ks * 32 + koffA;
        aq[ks].u[0] = *(const u32x4*)(p);
        aq[ks].u[1] = *(const u32x4*)(p + 16);
    }

    // ---------------- phase 1: scores + row max ----------------
    const __bf16* Kbase = Km + (size_t)bh * SEQ * DH;
    const float   scal  = 0.125f;  // 1/sqrt(64)
    for (int kc = 0; kc < SEQ; kc += 32) {
        v8f s[2];
#pragma unroll
        for (int nt = 0; nt < 2; ++nt)
#pragma unroll
            for (int e = 0; e < 8; ++e) s[nt][e] = 0.0f;
#pragma unroll
        for (int nt = 0; nt < 2; ++nt) {
            const __bf16* kp = Kbase + (size_t)(kc + nt * 16 + rsel) * DH;
#pragma unroll
            for (int ks = 0; ks < 2; ++ks) {
                Frag fb;
                fb.u[0] = *(const u32x4*)(kp + ks * 32 + koffB);
                fb.u[1] = *(const u32x4*)(kp + ks * 32 + koffB + 16);
                s[nt] = wmma_bf16(aq[ks].v, fb.v, s[nt]);
            }
        }
        // row max over this 32-key chunk, then fold into LDS stats
        float mred[8];
#pragma unroll
        for (int i = 0; i < 8; ++i) {
            float v = fmaxf(s[0][i], s[1][i]) * scal;
#pragma unroll
            for (int off = 1; off < 16; off <<= 1)
                v = fmaxf(v, __shfl_xor(v, off, 32));
            mred[i] = v;
        }
        if (lane == 0 || lane == 16) {
            int rb = qrb + ((lane == 16) ? 8 : 0);
#pragma unroll
            for (int i = 0; i < 8; ++i)
                rowmax[rb + i] = fmaxf(rowmax[rb + i], mred[i]);
        }
        // park scaled scores (bf16) in LDS
#pragma unroll
        for (int nt = 0; nt < 2; ++nt)
#pragma unroll
            for (int i = 0; i < 8; ++i) {
                int r = qrb + i + mhalf;
                int c = kc + nt * 16 + (lane & 15);
                sc[r * SC_STR + c] = (__bf16)(s[nt][i] * scal);
            }
    }
    __syncthreads();

    // ---------------- phase 2: softmax + P·V ----------------
    const float   mrow  = rowmax[qrb + (lane & 15)];
    const __bf16* Vbase = Vt + (size_t)bh * DH * SEQ;
    float lsum = 0.0f;
    v8f o[4];
#pragma unroll
    for (int nt = 0; nt < 4; ++nt)
#pragma unroll
        for (int e = 0; e < 8; ++e) o[nt][e] = 0.0f;

    for (int kc = 0; kc < SEQ; kc += 32) {
        const __bf16* sp = sc + (qrb + (lane & 15)) * SC_STR + kc + koffA;
        Frag praw, pf;
        praw.u[0] = *(const u32x4*)(sp);
        praw.u[1] = *(const u32x4*)(sp + 16);
        float part = 0.0f;
#pragma unroll
        for (int j = 0; j < 16; ++j) {
            float pv = __expf((float)praw.v[j] - mrow);
            part += pv;
            pf.v[j] = (__bf16)pv;
        }
        lsum += part + __shfl_xor(part, 16, 32);  // lane pair covers the row
#pragma unroll
        for (int nt = 0; nt < 4; ++nt) {
            const __bf16* vp = Vbase + (size_t)(nt * 16 + rsel) * SEQ + kc + koffB;
            Frag fv;
            fv.u[0] = *(const u32x4*)(vp);
            fv.u[1] = *(const u32x4*)(vp + 16);
            o[nt] = wmma_bf16(pf.v, fv.v, o[nt]);
        }
    }
    if (lane < 16) rowsum[qrb + lane] = lsum;
    __syncthreads();

    float inv[8];
#pragma unroll
    for (int i = 0; i < 8; ++i) inv[i] = 1.0f / rowsum[qrb + i + mhalf];
#pragma unroll
    for (int nt = 0; nt < 4; ++nt)
#pragma unroll
        for (int i = 0; i < 8; ++i) {
            int m   = b * SEQ + qb * 64 + qrb + i + mhalf;
            int col = h * 64 + nt * 16 + (lane & 15);
            ctx[(size_t)m * EMB + col] = (__bf16)(o[nt][i] * inv[i]);
        }
}

// ---------------------------------------------------------------------------
// LayerNorm over E=512; one row per block, 256 threads x 2 elements.
// ---------------------------------------------------------------------------
template <bool WB16>
__global__ __launch_bounds__(256) void layernorm_kernel(
    const float* __restrict__ x, const float* __restrict__ g,
    const float* __restrict__ be, float* __restrict__ yf,
    __bf16* __restrict__ yb) {
    const int row  = blockIdx.x;
    const int tid  = threadIdx.x;
    const int lane = tid & 31;
    const int wv   = tid >> 5;
    const float* xr = x + (size_t)row * EMB;
    float2 v = *(const float2*)(xr + tid * 2);

    float s = v.x + v.y;
#pragma unroll
    for (int off = 1; off < 32; off <<= 1) s += __shfl_xor(s, off, 32);
    __shared__ float rb[8];
    if (lane == 0) rb[wv] = s;
    __syncthreads();
    float tot = 0.0f;
#pragma unroll
    for (int i = 0; i < 8; ++i) tot += rb[i];
    const float mu = tot * (1.0f / EMB);

    float d0 = v.x - mu, d1 = v.y - mu;
    float q = d0 * d0 + d1 * d1;
#pragma unroll
    for (int off = 1; off < 32; off <<= 1) q += __shfl_xor(q, off, 32);
    __syncthreads();
    if (lane == 0) rb[wv] = q;
    __syncthreads();
    float qt = 0.0f;
#pragma unroll
    for (int i = 0; i < 8; ++i) qt += rb[i];
    const float r = rsqrtf(qt * (1.0f / EMB) + 1e-5f);

    const int c0 = tid * 2;
    float y0 = d0 * r * g[c0] + be[c0];
    float y1 = d1 * r * g[c0 + 1] + be[c0 + 1];
    float2 yo; yo.x = y0; yo.y = y1;
    *(float2*)(yf + (size_t)row * EMB + c0) = yo;
    if constexpr (WB16) {
        yb[(size_t)row * EMB + c0]     = (__bf16)y0;
        yb[(size_t)row * EMB + c0 + 1] = (__bf16)y1;
    }
}

// ---------------------------------------------------------------------------
// host-side orchestration
// ---------------------------------------------------------------------------
extern "C" void kernel_launch(void* const* d_in, const int* in_sizes, int n_in,
                              void* d_out, int out_size, void* d_ws, size_t ws_size,
                              hipStream_t stream) {
    const float* query = (const float*)d_in[0];
    const float* key_t = (const float*)d_in[1];
    const float* value = (const float*)d_in[2];
    const float* Wq = (const float*)d_in[3];  const float* bq = (const float*)d_in[4];
    const float* Wk = (const float*)d_in[5];  const float* bk = (const float*)d_in[6];
    const float* Wv = (const float*)d_in[7];  const float* bv = (const float*)d_in[8];
    const float* Wo = (const float*)d_in[9];  const float* bo = (const float*)d_in[10];
    const float* g1 = (const float*)d_in[11]; const float* be1 = (const float*)d_in[12];
    const float* g2 = (const float*)d_in[13]; const float* be2 = (const float*)d_in[14];
    const float* W1 = (const float*)d_in[15]; const float* b1 = (const float*)d_in[16];
    const float* W2 = (const float*)d_in[17]; const float* b2 = (const float*)d_in[18];

    char* ws = (char*)d_ws;
    size_t off = 0;
    auto carve = [&](size_t bytes) -> void* {
        void* p = ws + off;
        off += (bytes + 255) & ~(size_t)255;
        return p;
    };
    const size_t ME = (size_t)MROWS * EMB;
    __bf16* Xq  = (__bf16*)carve(ME * 2);
    __bf16* Xk  = (__bf16*)carve(ME * 2);
    __bf16* Xv  = (__bf16*)carve(ME * 2);
    __bf16* WqT = (__bf16*)carve((size_t)EMB * EMB * 2);
    __bf16* WkT = (__bf16*)carve((size_t)EMB * EMB * 2);
    __bf16* WvT = (__bf16*)carve((size_t)EMB * EMB * 2);
    __bf16* WoT = (__bf16*)carve((size_t)EMB * EMB * 2);
    __bf16* W1T = (__bf16*)carve((size_t)EMB * FFN * 2);  // [FFN,EMB]
    __bf16* W2T = (__bf16*)carve((size_t)FFN * EMB * 2);  // [EMB,FFN]
    __bf16* Qb  = (__bf16*)carve(ME * 2);  // [B,H,S,DH]
    __bf16* Kb  = (__bf16*)carve(ME * 2);  // [B,H,S,DH]
    __bf16* Vtb = (__bf16*)carve(ME * 2);  // [B,H,DH,S]
    __bf16* Ctx = (__bf16*)carve(ME * 2);  // [M,EMB]
    float*  T1  = (float*)carve(ME * 4);   // attn_out + residual
    float*  L1f = (float*)carve(ME * 4);   // LN1 fp32
    __bf16* L1b = (__bf16*)carve(ME * 2);  // LN1 bf16 (FFN input)
    __bf16* F1  = (__bf16*)carve((size_t)MROWS * FFN * 2);
    float*  T2  = (float*)carve(ME * 4);   // ffn_out + residual

    // 1) convert activations + weights
    cvt_bf16_kernel<<<1024, 256, 0, stream>>>(query, Xq, (int)ME);
    cvt_bf16_kernel<<<1024, 256, 0, stream>>>(key_t, Xk, (int)ME);
    cvt_bf16_kernel<<<1024, 256, 0, stream>>>(value, Xv, (int)ME);
    cvtT_bf16_kernel<<<512, 256, 0, stream>>>(Wq, WqT, EMB, EMB);
    cvtT_bf16_kernel<<<512, 256, 0, stream>>>(Wk, WkT, EMB, EMB);
    cvtT_bf16_kernel<<<512, 256, 0, stream>>>(Wv, WvT, EMB, EMB);
    cvtT_bf16_kernel<<<512, 256, 0, stream>>>(Wo, WoT, EMB, EMB);
    cvtT_bf16_kernel<<<1024, 256, 0, stream>>>(W1, W1T, EMB, FFN);
    cvtT_bf16_kernel<<<1024, 256, 0, stream>>>(W2, W2T, FFN, EMB);

    const size_t gsmem = 4 * TILEB;    // double-buffered A+B (40 KB)
    dim3 gP(EMB / 128, MROWS / 128);   // 4 x 64
    dim3 gF1(FFN / 128, MROWS / 128);  // 16 x 64

    // 2) QKV projections (WMMA)
    gemm_kernel<EP_QK><<<gP, 256, gsmem, stream>>>(Xq, WqT, bq, Qb, nullptr, MROWS, EMB, EMB);
    gemm_kernel<EP_QK><<<gP, 256, gsmem, stream>>>(Xk, WkT, bk, Kb, nullptr, MROWS, EMB, EMB);
    gemm_kernel<EP_V><<<gP, 256, gsmem, stream>>>(Xv, WvT, bv, Vtb, nullptr, MROWS, EMB, EMB);

    // 3) attention (WMMA, big-LDS score tile)
    const size_t asmem = (size_t)64 * (SEQ + 8) * 2 + (size_t)64 * 72 * 2 + 2 * 64 * 4;
    attention_kernel<<<BATCH * NH * (SEQ / 64), 128, asmem, stream>>>(Qb, Kb, Vtb, Ctx);

    // 4) output projection + residual, LN1
    gemm_kernel<EP_O><<<gP, 256, gsmem, stream>>>(Ctx, WoT, bo, T1, query, MROWS, EMB, EMB);
    layernorm_kernel<true><<<MROWS, 256, 0, stream>>>(T1, g1, be1, L1f, L1b);

    // 5) FFN (GELU) + residual, LN2 -> d_out
    gemm_kernel<EP_F1><<<gF1, 256, gsmem, stream>>>(L1b, W1T, b1, F1, nullptr, MROWS, FFN, EMB);
    gemm_kernel<EP_F2><<<gP, 256, gsmem, stream>>>(F1, W2T, b2, T2, L1f, MROWS, EMB, FFN);
    layernorm_kernel<false><<<MROWS, 256, 0, stream>>>(T2, g2, be2, (float*)d_out, nullptr);

    (void)in_sizes; (void)n_in; (void)out_size; (void)ws_size;
}
